// SelfAttention_40235253629610
// MI455X (gfx1250) — matrix operand order repaired and checked by reading, not compiled, not run on hardware
//
#include <hip/hip_runtime.h>
#include <math.h>

// ---------------- types ----------------
typedef _Float16 h8  __attribute__((ext_vector_type(8)));
typedef _Float16 h16 __attribute__((ext_vector_type(16)));
typedef float    f8  __attribute__((ext_vector_type(8)));
typedef float    f4  __attribute__((ext_vector_type(4)));

#define B_   2
#define L_   2048
#define D_   512
#define H_   8
#define DK_  64
#define DFF_ 1024
#define NL_  2

__device__ __forceinline__ h16 cat8(h8 lo, h8 hi) {
  return __builtin_shufflevector(lo, hi, 0,1,2,3,4,5,6,7,8,9,10,11,12,13,14,15);
}
__device__ __forceinline__ f8 wmma16(h16 a, h16 b, f8 c) {
  // D = A(16x32 f16) * B(32x16 f16) + C(16x16 f32)
  return __builtin_amdgcn_wmma_f32_16x16x32_f16(false, a, false, b, (short)0, c, false, false);
}

// ---------------- generic WMMA GEMM: C[M,N] = A[M,K] * Wt[N,K]^T ----------------
// Wave tile: 32(M) x 64(N) -> 8 wmma per 32-deep k-step from 2 A-frags + 4 B-frags.
// flags: 1=add bias[n], 2=relu, 4=write f32, 8=write f16
__global__ __launch_bounds__(256) void gemm_kernel(
    const _Float16* __restrict__ A, const _Float16* __restrict__ Wt,
    const float* __restrict__ bias, float* __restrict__ C32,
    _Float16* __restrict__ C16, int M, int N, int K, int flags)
{
  const int lane = threadIdx.x & 31;
  const int wv   = threadIdx.x >> 5;
  const int hs   = lane >> 4;      // half-wave select
  const int cq   = lane & 15;
  const int mbase = (blockIdx.x * 8 + wv) * 32;  // 32 rows per wave
  const int n0 = blockIdx.y * 64;                // 64 cols per wave
  const _Float16* arow0 = A + (size_t)(mbase + cq) * K;
  const _Float16* arow1 = A + (size_t)(mbase + 16 + cq) * K;

  f8 acc[2][4] = {};
  for (int k0 = 0; k0 < K; k0 += 32) {
    // A fragments: lane row = cq; element e -> k = 8*hs + e + (e>=8 ? 8 : 0)
    h16 a0 = cat8(*(const h8*)(arow0 + k0 + hs * 8),
                  *(const h8*)(arow0 + k0 + hs * 8 + 16));
    h16 a1 = cat8(*(const h8*)(arow1 + k0 + hs * 8),
                  *(const h8*)(arow1 + k0 + hs * 8 + 16));
#pragma unroll
    for (int t = 0; t < 4; ++t) {
      // B fragment from Wt row n: element e -> k = 8*hs + e + (e>=8 ? 8 : 0)
      const _Float16* wrow = Wt + (size_t)(n0 + t * 16 + cq) * K + k0 + hs * 8;
      h16 bf = cat8(*(const h8*)wrow, *(const h8*)(wrow + 16));
      acc[0][t] = wmma16(a0, bf, acc[0][t]);
      acc[1][t] = wmma16(a1, bf, acc[1][t]);
    }
  }
#pragma unroll
  for (int ms = 0; ms < 2; ++ms) {
    const int m0 = mbase + ms * 16 + hs * 8;
#pragma unroll
    for (int t = 0; t < 4; ++t) {
      const int n = n0 + t * 16 + cq;
      float bv = (flags & 1) ? bias[n] : 0.0f;
#pragma unroll
      for (int r = 0; r < 8; ++r) {
        float v = acc[ms][t][r] + bv;
        if (flags & 2) v = fmaxf(v, 0.0f);
        size_t off = (size_t)(m0 + r) * N + n;
        if (flags & 4) C32[off] = v;
        if (flags & 8) C16[off] = (_Float16)v;
      }
    }
  }
}

// ---------------- flash attention (full, non-causal) ----------------
// qkv: (B*L, 1536) f16  [q | k | v], per-head cols h*64+dk
// vT : (B*H, 64, L) f16
// out: (B*L, 512) f32 concat-head attention output
__global__ __launch_bounds__(128) void attn_kernel(
    const _Float16* __restrict__ qkv, const _Float16* __restrict__ vT,
    float* __restrict__ out)
{
  __shared__ _Float16 plds[4][16][32];      // per-wave P tile [q][m], 4KB total
  const int lane = threadIdx.x & 31;
  const int wv   = threadIdx.x >> 5;
  const int hs   = lane >> 4;
  const int cq   = lane & 15;               // query column within wave tile
  const int bh = blockIdx.y;
  const int b = bh >> 3, h = bh & 7;
  const int qbase = blockIdx.x * 64 + wv * 16;
  const float scale = 0.04419417382415922f; // 1/sqrt(512)

  // Q^T B-fragments (dk x q), loaded once: element e -> dk = base + 8*hs + e + (e>=8 ? 8 : 0)
  const _Float16* qrow = qkv + (size_t)(b * L_ + qbase + cq) * (3 * D_) + h * DK_;
  h16 bq0 = cat8(*(const h8*)(qrow + hs * 8), *(const h8*)(qrow + hs * 8 + 16));        // dk 0..31
  h16 bq1 = cat8(*(const h8*)(qrow + 32 + hs * 8), *(const h8*)(qrow + 32 + hs * 8 + 16));   // dk 32..63

  const _Float16* kbuf = qkv + (size_t)b * L_ * (3 * D_) + D_ + h * DK_;
  const _Float16* vbuf = vT + (size_t)bh * DK_ * L_;
  const int ka = hs * 8;

  f8 o[4] = {};                 // out^T tiles: (dv = 16*t + 8*hs + r, q = cq)
  float mrun = -3.0e38f, srun = 0.0f;

  for (int kb = 0; kb < L_; kb += 32) {
    // K A-fragments (16 keys x 32 dk), two key tiles, two dk halves
    const _Float16* kr0 = kbuf + (size_t)(kb + cq) * (3 * D_);
    const _Float16* kr1 = kbuf + (size_t)(kb + 16 + cq) * (3 * D_);
    h16 a00 = cat8(*(const h8*)(kr0 + ka),      *(const h8*)(kr0 + ka + 16));
    h16 a01 = cat8(*(const h8*)(kr0 + 32 + ka), *(const h8*)(kr0 + 32 + ka + 16));
    h16 a10 = cat8(*(const h8*)(kr1 + ka),      *(const h8*)(kr1 + ka + 16));
    h16 a11 = cat8(*(const h8*)(kr1 + 32 + ka), *(const h8*)(kr1 + 32 + ka + 16));

    f8 s0 = {}, s1 = {};                       // S^T tiles: (m, q)
    s0 = wmma16(a00, bq0, s0);
    s0 = wmma16(a01, bq1, s0);
    s1 = wmma16(a10, bq0, s1);
    s1 = wmma16(a11, bq1, s1);

    float bm = -3.0e38f;
#pragma unroll
    for (int r = 0; r < 8; ++r) {
      s0[r] *= scale; s1[r] *= scale;
      bm = fmaxf(bm, fmaxf(s0[r], s1[r]));
    }
    bm = fmaxf(bm, __shfl_xor(bm, 16, 32));    // combine the two m-halves
    float mnew = fmaxf(mrun, bm);
    float corr = __expf(mrun - mnew);

    h8 p0 = {}, p1 = {};
    float ls = 0.0f;
#pragma unroll
    for (int r = 0; r < 8; ++r) {
      float e0 = __expf(s0[r] - mnew);
      float e1 = __expf(s1[r] - mnew);
      ls += e0 + e1;
      p0[r] = (_Float16)e0;
      p1[r] = (_Float16)e1;
    }
    ls += __shfl_xor(ls, 16, 32);
    srun = srun * corr + ls;
    mrun = mnew;
#pragma unroll
    for (int t = 0; t < 4; ++t) o[t] *= corr;  // per-lane uniform (q = cq)

    // relayout P (D-layout) -> B-fragment via per-wave LDS tile [q][m]
    _Float16* lq = &plds[wv][cq][0];
    *(h8*)(lq + hs * 8)      = p0;             // m = 8*hs + r
    *(h8*)(lq + 16 + hs * 8) = p1;             // m = 16 + 8*hs + r
    h16 bp = cat8(*(const h8*)(&plds[wv][cq][hs * 8]), *(const h8*)(&plds[wv][cq][16 + hs * 8]));  // element e -> m = 8*hs + e + (e>=8 ? 8 : 0)

    // out^T += V^T (16dv x 32m) * P^T (32m x 16q)
#pragma unroll
    for (int t = 0; t < 4; ++t) {
      const _Float16* vr = vbuf + (size_t)(t * 16 + cq) * L_ + kb;
      h16 av = cat8(*(const h8*)(vr + ka), *(const h8*)(vr + ka + 16));
      o[t] = wmma16(av, bp, o[t]);
    }
  }

  float inv = 1.0f / srun;
  float* orow = out + (size_t)(b * L_ + qbase + cq) * D_ + h * DK_ + hs * 8;
#pragma unroll
  for (int t = 0; t < 4; ++t) {
    f4 lo = { o[t][0] * inv, o[t][1] * inv, o[t][2] * inv, o[t][3] * inv };
    f4 hi = { o[t][4] * inv, o[t][5] * inv, o[t][6] * inv, o[t][7] * inv };
    *(f4*)(orow + t * 16)     = lo;
    *(f4*)(orow + t * 16 + 4) = hi;
  }
}

// ---------------- LayerNorm (torch.std ddof=1, /(sigma+eps)) with fused residual ----------------
__global__ __launch_bounds__(256) void ln_kernel(
    const float* __restrict__ za, const float* __restrict__ zb,
    const float* __restrict__ gamma, const float* __restrict__ beta,
    float* __restrict__ out32, _Float16* __restrict__ out16)
{
  __shared__ float red[2][8];
  const int row = blockIdx.x;
  const int tid = threadIdx.x;
  const size_t base = (size_t)row * D_;
  float v0 = za[base + tid]       + zb[base + tid];
  float v1 = za[base + tid + 256] + zb[base + tid + 256];
  float s  = v0 + v1;
  float sq = v0 * v0 + v1 * v1;
#pragma unroll
  for (int off = 16; off > 0; off >>= 1) {
    s  += __shfl_xor(s,  off, 32);
    sq += __shfl_xor(sq, off, 32);
  }
  if ((tid & 31) == 0) { red[0][tid >> 5] = s; red[1][tid >> 5] = sq; }
  __syncthreads();
  float ts = 0.f, tq = 0.f;
#pragma unroll
  for (int i = 0; i < 8; ++i) { ts += red[0][i]; tq += red[1][i]; }
  float mu  = ts * (1.0f / (float)D_);
  float var = fmaxf((tq - mu * mu * (float)D_) * (1.0f / (float)(D_ - 1)), 0.0f);
  float inv = 1.0f / (sqrtf(var) + 1e-3f);
  float y0 = (v0 - mu) * inv * gamma[tid]       + beta[tid];
  float y1 = (v1 - mu) * inv * gamma[tid + 256] + beta[tid + 256];
  out32[base + tid]       = y0;
  out32[base + tid + 256] = y1;
  out16[base + tid]       = (_Float16)y0;
  out16[base + tid + 256] = (_Float16)y1;
}

// ---------------- helpers ----------------
__global__ void addpos_kernel(const float* __restrict__ x, const float* __restrict__ pos,
                              float* __restrict__ o32, _Float16* __restrict__ o16)
{
  size_t i = (size_t)blockIdx.x * 256 + threadIdx.x;     // B*L*D
  float v = x[i] + pos[i & (size_t)(L_ * D_ - 1)];
  o32[i] = v;
  o16[i] = (_Float16)v;
}

__global__ void transpose_v_kernel(const _Float16* __restrict__ qkv, _Float16* __restrict__ vT)
{
  int i = blockIdx.x * 256 + threadIdx.x;                // B*H*64*L
  int m  = i & (L_ - 1);
  int t  = i >> 11;
  int dv = t & (DK_ - 1);
  int bh = t >> 6;
  int b = bh >> 3, h = bh & (H_ - 1);
  vT[i] = qkv[(size_t)(b * L_ + m) * (3 * D_) + 2 * D_ + h * DK_ + dv];
}

// wq/wk/wv: (NL, H, D, DK) f32 -> wqkv_t: (NL, 1536, 512) f16, row n = which*512+h*64+dk, col d
__global__ void prep_qkv_kernel(const float* __restrict__ wq, const float* __restrict__ wk,
                                const float* __restrict__ wvp, _Float16* __restrict__ outw)
{
  int i = blockIdx.x * 256 + threadIdx.x;                // NL*1536*512
  int d = i & (D_ - 1);
  int r = i >> 9;
  int n = r % 1536;
  int layer = r / 1536;
  int which = n >> 9;
  int h  = (n >> 6) & 7;
  int dk = n & 63;
  const float* src = (which == 0) ? wq : (which == 1) ? wk : wvp;
  outw[i] = (_Float16)src[(((size_t)layer * H_ + h) * D_ + d) * DK_ + dk];
}

__global__ void cvt16_kernel(const float* __restrict__ in, _Float16* __restrict__ outh)
{
  int i = blockIdx.x * 256 + threadIdx.x;
  outh[i] = (_Float16)in[i];
}

// ---------------- launch ----------------
extern "C" void kernel_launch(void* const* d_in, const int* in_sizes, int n_in,
                              void* d_out, int out_size, void* d_ws, size_t ws_size,
                              hipStream_t stream) {
  const float* x     = (const float*)d_in[0];
  const float* pos   = (const float*)d_in[1];
  const float* wq    = (const float*)d_in[2];
  const float* wk    = (const float*)d_in[3];
  const float* wv    = (const float*)d_in[4];
  const float* ln1a  = (const float*)d_in[5];
  const float* ln1b  = (const float*)d_in[6];
  const float* w1    = (const float*)d_in[7];
  const float* b1    = (const float*)d_in[8];
  const float* w2    = (const float*)d_in[9];
  const float* b2    = (const float*)d_in[10];
  const float* ln2a  = (const float*)d_in[11];
  const float* ln2b  = (const float*)d_in[12];
  float* outp = (float*)d_out;

  char* ws = (char*)d_ws;
  size_t off = 0;
  auto take = [&](size_t bytes) -> void* {
    void* p = ws + off;
    off = (off + bytes + 255) & ~(size_t)255;
    return p;
  };
  const size_t BLD = (size_t)B_ * L_ * D_;               // 2,097,152
  float*     xa32   = (float*)take(BLD * 4);
  _Float16*  xa16   = (_Float16*)take(BLD * 2);
  float*     xb32   = (float*)take(BLD * 4);
  _Float16*  xb16   = (_Float16*)take(BLD * 2);
  _Float16*  qkv16  = (_Float16*)take((size_t)B_ * L_ * 3 * D_ * 2);
  _Float16*  vT16   = (_Float16*)take((size_t)B_ * H_ * DK_ * L_ * 2);
  float*     attn32 = (float*)take(BLD * 4);
  _Float16*  ff16   = (_Float16*)take((size_t)B_ * L_ * DFF_ * 2);
  float*     ff2    = (float*)take(BLD * 4);
  _Float16*  wqkv16 = (_Float16*)take((size_t)NL_ * 3 * D_ * D_ * 2);
  _Float16*  w1h    = (_Float16*)take((size_t)NL_ * DFF_ * D_ * 2);
  _Float16*  w2h    = (_Float16*)take((size_t)NL_ * D_ * DFF_ * 2);

  // weight prep (f32 -> f16, QKV also transposed/interleaved to (N,K))
  prep_qkv_kernel<<<(NL_ * 3 * D_ * D_) / 256, 256, 0, stream>>>(wq, wk, wv, wqkv16);
  cvt16_kernel<<<(NL_ * DFF_ * D_) / 256, 256, 0, stream>>>(w1, w1h);
  cvt16_kernel<<<(NL_ * D_ * DFF_) / 256, 256, 0, stream>>>(w2, w2h);

  // x = x + pos
  addpos_kernel<<<(int)(BLD / 256), 256, 0, stream>>>(x, pos, xa32, xa16);

  const int M = B_ * L_;                                  // 4096
  for (int l = 0; l < NL_; ++l) {
    // fused QKV projection: (4096,512) x (1536,512)^T -> f16 qkv
    gemm_kernel<<<dim3(M / 256, (3 * D_) / 64), 256, 0, stream>>>(
        xa16, wqkv16 + (size_t)l * 3 * D_ * D_, nullptr, nullptr, qkv16,
        M, 3 * D_, D_, /*OUT16*/8);
    transpose_v_kernel<<<(B_ * H_ * DK_ * L_) / 256, 256, 0, stream>>>(qkv16, vT16);
    attn_kernel<<<dim3(L_ / 64, B_ * H_), 128, 0, stream>>>(qkv16, vT16, attn32);
    ln_kernel<<<M, 256, 0, stream>>>(attn32, xa32, ln1a + l * D_, ln1b + l * D_, xb32, xb16);
    // FFN1: relu(x @ w1^T + b1) -> f16
    gemm_kernel<<<dim3(M / 256, DFF_ / 64), 256, 0, stream>>>(
        xb16, w1h + (size_t)l * DFF_ * D_, b1 + l * DFF_, nullptr, ff16,
        M, DFF_, D_, 1 | 2 | 8);
    // FFN2: ff @ w2^T + b2 -> f32
    gemm_kernel<<<dim3(M / 256, D_ / 64), 256, 0, stream>>>(
        ff16, w2h + (size_t)l * D_ * DFF_, b2 + l * D_, ff2, nullptr,
        M, D_, DFF_, 1 | 4);
    float* o32 = (l == NL_ - 1) ? outp : xa32;
    ln_kernel<<<M, 256, 0, stream>>>(ff2, xb32, ln2a + l * D_, ln2b + l * D_, o32, xa16);
  }
}
